// LongformerEncoder_29901562314911
// MI455X (gfx1250) — compile-verified, matrix operand order checked by reading, not run on hardware
//
#include <hip/hip_runtime.h>

// ---------------------------------------------------------------------------
// Types for CDNA5 WMMA (wave32, 16x16x32 bf16 -> f32)
// ---------------------------------------------------------------------------
typedef __attribute__((ext_vector_type(16))) __bf16 bf16x16;
typedef __attribute__((ext_vector_type(8)))  __bf16 bf16x8;
typedef __attribute__((ext_vector_type(8)))  float  v8f;
typedef __attribute__((ext_vector_type(4)))  int    v4i;

union BF16x16U { bf16x16 v; struct { bf16x8 lo, hi; } s; };
union BF8U     { float4 f4; __bf16 h[8]; };

// Fragment load matching ISA 16-bit A/B layout: per lane the two 8-element
// K-groups live at +0 and +16 elements (caller adds (lane>>4)*8).
__device__ __forceinline__ bf16x16 frag_ld(const __bf16* p) {
  BF16x16U u;
  u.s.lo = *(const bf16x8*)(p);
  u.s.hi = *(const bf16x8*)(p + 16);
  return u.v;
}

__device__ __forceinline__ v8f wmma_bf16(bf16x16 a, bf16x16 b, v8f c) {
  // (neg_a, A, neg_b, B, c_mod, C, reuse_a, reuse_b)
  return __builtin_amdgcn_wmma_f32_16x16x32_bf16(false, a, false, b, (short)0, c,
                                                 false, false);
}

// ---------------------------------------------------------------------------
// CDNA5 async global->LDS copy (ASYNCcnt-tracked DMA), with safe fallback.
// global_load_async_to_lds_b128 stages 16B per active lane without touching
// VGPR data paths; we overlap it with WMMA work and fence with
// s_wait_asynccnt before the workgroup barrier.
// Builtin signature (from hipcc diagnostic): arg0 = v4i addrspace(1)*,
// arg1 = v4i addrspace(3)*, then immediate offset and cpol.
// ---------------------------------------------------------------------------
#if __has_builtin(__builtin_amdgcn_global_load_async_to_lds_b128) && \
    __has_builtin(__builtin_amdgcn_s_wait_asynccnt)
#define HAVE_ASYNC_LDS 1
#else
#define HAVE_ASYNC_LDS 0
#endif

__device__ __forceinline__ void cp_async16(void* lds, const void* gmem) {
#if HAVE_ASYNC_LDS
  __builtin_amdgcn_global_load_async_to_lds_b128(
      (__attribute__((address_space(1))) v4i*)(gmem),
      (__attribute__((address_space(3))) v4i*)(lds), 0, 0);
#else
  *(float4*)lds = *(const float4*)gmem;
#endif
}

__device__ __forceinline__ void async_fence() {
#if HAVE_ASYNC_LDS
  __builtin_amdgcn_s_wait_asynccnt(0);
#endif
}

// ---------------------------------------------------------------------------
// Problem constants
// ---------------------------------------------------------------------------
#define CB   4
#define CS   2048
#define CD   768
#define CDFF 3072
#define CW   256
#define CBS  (CB * CS)   // 8192 rows

// ---------------------------------------------------------------------------
// fp32 -> bf16 conversion
// ---------------------------------------------------------------------------
__global__ __launch_bounds__(256) void k_cvt(const float* __restrict__ in,
                                             __bf16* __restrict__ out, int n) {
  int i = blockIdx.x * 256 + threadIdx.x;
  if (i < n) out[i] = (__bf16)in[i];
}

// ---------------------------------------------------------------------------
// Block reduction (256 threads = 8 wave32)
// ---------------------------------------------------------------------------
__device__ __forceinline__ float block_sum(float val, float* sb) {
  #pragma unroll
  for (int o = 16; o > 0; o >>= 1) val += __shfl_xor(val, o, 32);
  int w = threadIdx.x >> 5;
  if ((threadIdx.x & 31) == 0) sb[w] = val;
  __syncthreads();
  float r = (threadIdx.x < 8) ? sb[threadIdx.x] : 0.0f;
  if (threadIdx.x < 32) {
    #pragma unroll
    for (int o = 4; o > 0; o >>= 1) r += __shfl_xor(r, o, 32);
  }
  if (threadIdx.x == 0) sb[8] = r;
  __syncthreads();
  float out = sb[8];
  __syncthreads();
  return out;
}

// ---------------------------------------------------------------------------
// h = LayerNorm(emb[x] + pos); one block per token, 256 threads x 3 elems
// ---------------------------------------------------------------------------
__global__ __launch_bounds__(256) void k_embed_ln(
    const int* __restrict__ x, const float* __restrict__ emb,
    const float* __restrict__ pos, const float* __restrict__ g,
    const float* __restrict__ bet, float* __restrict__ hf,
    __bf16* __restrict__ hb) {
  __shared__ float sb[16];
  const int tok = blockIdx.x;
  const int s   = tok & (CS - 1);
  const int id  = x[tok];
  float v[3];
  #pragma unroll
  for (int i = 0; i < 3; ++i) {
    int d = threadIdx.x + i * 256;
    v[i] = emb[(size_t)id * CD + d] + pos[(size_t)s * CD + d];
  }
  float mean = block_sum(v[0] + v[1] + v[2], sb) * (1.0f / CD);
  float var = 0.0f;
  #pragma unroll
  for (int i = 0; i < 3; ++i) { float t = v[i] - mean; var += t * t; }
  var = block_sum(var, sb) * (1.0f / CD);
  float rs = rsqrtf(var + 1e-5f);
  #pragma unroll
  for (int i = 0; i < 3; ++i) {
    int d = threadIdx.x + i * 256;
    float y = (v[i] - mean) * rs * g[d] + bet[d];
    hf[(size_t)tok * CD + d] = y;
    hb[(size_t)tok * CD + d] = (__bf16)y;
  }
}

// ---------------------------------------------------------------------------
// y = LayerNorm(a + r); optional bf16 mirror
// ---------------------------------------------------------------------------
__global__ __launch_bounds__(256) void k_add_ln(
    const float* __restrict__ a, const float* __restrict__ r,
    const float* __restrict__ g, const float* __restrict__ bet,
    float* __restrict__ of, __bf16* __restrict__ ob) {
  __shared__ float sb[16];
  const size_t tok = blockIdx.x;
  float v[3];
  #pragma unroll
  for (int i = 0; i < 3; ++i) {
    int d = threadIdx.x + i * 256;
    v[i] = a[tok * CD + d] + r[tok * CD + d];
  }
  float mean = block_sum(v[0] + v[1] + v[2], sb) * (1.0f / CD);
  float var = 0.0f;
  #pragma unroll
  for (int i = 0; i < 3; ++i) { float t = v[i] - mean; var += t * t; }
  var = block_sum(var, sb) * (1.0f / CD);
  float rs = rsqrtf(var + 1e-5f);
  #pragma unroll
  for (int i = 0; i < 3; ++i) {
    int d = threadIdx.x + i * 256;
    float y = (v[i] - mean) * rs * g[d] + bet[d];
    of[tok * CD + d] = y;
    if (ob) ob[tok * CD + d] = (__bf16)y;
  }
}

// ---------------------------------------------------------------------------
// Tiled WMMA GEMM: C[M,N] = A[M,K] * B[K,N] (bf16 in, f32 accumulate)
// Block 256 threads (8 waves), tile 128 x BNT x 32, double-buffered LDS.
// Wave grid 2x4: each wave computes 64 x (BNT/4) -> 4 x (BNT/64) WMMA tiles.
//   BNT=256: 16 WMMA per k-step per wave (high matrix-op density)
//   BNT=128:  8 WMMA per k-step per wave (keeps grid occupancy for N=768)
// A-tile staged via async global->LDS DMA; B-tile staged transposed via VALU.
// fuse: 0 = none, 1 = tanh-GELU
// ---------------------------------------------------------------------------
template <int BNT>
__global__ __launch_bounds__(256) void k_gemm_bf16(
    const __bf16* __restrict__ A, const __bf16* __restrict__ Bw,
    float* __restrict__ C, __bf16* __restrict__ Cb,
    int M, int N, int K, int fuse) {
  constexpr int NTC = BNT / 64;            // n-tiles per wave
  __shared__ __bf16 As[2][128 * 32];       // [m][k], ld=32
  __shared__ __bf16 Bs[2][BNT * 32];       // [n][k], ld=32 (transposed stage)

  const int tid  = threadIdx.x;
  const int lane = tid & 31;
  const int wave = tid >> 5;
  const int wm = wave >> 2, wn = wave & 3;
  const int lh = lane >> 4, l16 = lane & 15;
  const int row0 = blockIdx.y * 128, col0 = blockIdx.x * BNT;

  v8f acc[4][NTC];
  #pragma unroll
  for (int i = 0; i < 4; ++i)
    #pragma unroll
    for (int j = 0; j < NTC; ++j) acc[i][j] = v8f{};

  auto stage = [&](int buf, int k0) {
    #pragma unroll
    for (int t = 0; t < 2; ++t) {          // A tile: 512 x 16B chunks
      int c = tid + t * 256;
      int am = c >> 2, ak = (c & 3) << 3;
      cp_async16(&As[buf][am * 32 + ak],
                 A + (size_t)(row0 + am) * K + k0 + ak);
    }
    #pragma unroll
    for (int t = 0; t < BNT / 64; ++t) {   // B tile: transpose into [n][k]
      int c = tid + t * 256;
      int bk = c / (BNT / 8), bn = (c % (BNT / 8)) << 3;
      BF8U u;
      u.f4 = *(const float4*)(Bw + (size_t)(k0 + bk) * N + col0 + bn);
      #pragma unroll
      for (int i = 0; i < 8; ++i) Bs[buf][(bn + i) * 32 + bk] = u.h[i];
    }
  };

  stage(0, 0);
  async_fence();
  __syncthreads();
  const int nk = K >> 5;
  for (int ks = 0; ks < nk; ++ks) {
    const int cur = ks & 1;
    if (ks + 1 < nk) {
      if (ks + 2 < nk)   // speculative L2 prefetch of the tile after next
        __builtin_prefetch(A + (size_t)(row0 + (tid >> 2)) * K + (ks + 2) * 32, 0, 1);
      stage(cur ^ 1, (ks + 1) * 32);
    }
    bf16x16 af[4], bfg[NTC];
    #pragma unroll
    for (int mt = 0; mt < 4; ++mt)
      af[mt] = frag_ld(&As[cur][(wm * 64 + mt * 16 + l16) * 32 + lh * 8]);
    #pragma unroll
    for (int nt = 0; nt < NTC; ++nt)
      bfg[nt] = frag_ld(&Bs[cur][(wn * (BNT / 4) + nt * 16 + l16) * 32 + lh * 8]);
    #pragma unroll
    for (int mt = 0; mt < 4; ++mt)
      #pragma unroll
      for (int nt = 0; nt < NTC; ++nt)
        acc[mt][nt] = wmma_bf16(af[mt], bfg[nt], acc[mt][nt]);
    async_fence();     // next-buffer DMA must land before the barrier
    __syncthreads();
  }

  // Epilogue: C/D layout -> row = r + 8*(lane>>4), col = lane&15
  #pragma unroll
  for (int mt = 0; mt < 4; ++mt) {
    #pragma unroll
    for (int nt = 0; nt < NTC; ++nt) {
      #pragma unroll
      for (int r = 0; r < 8; ++r) {
        int row = row0 + wm * 64 + mt * 16 + lh * 8 + r;
        int col = col0 + wn * (BNT / 4) + nt * 16 + l16;
        float vv = acc[mt][nt][r];
        if (fuse == 1) {
          float x3 = vv * vv * vv;
          vv = 0.5f * vv * (1.0f + tanhf(0.7978845608028654f * (vv + 0.044715f * x3)));
        }
        size_t idx = (size_t)row * N + col;
        if (C)  C[idx]  = vv;
        if (Cb) Cb[idx] = (__bf16)vv;
      }
    }
  }
}

// ---------------------------------------------------------------------------
// Fused flash-style attention, templated on per-head dim and mask mode.
//   MODE 0: full attention with key-padding mask (x[b,k] != 0)   [outer MHA]
//   MODE 1: sliding-window band mask |k - q| <= CW               [longformer]
// Block = 128 threads (4 waves), handles one (b, head, 16-query tile).
// Key chunks of 64; wave w computes key tile w of the score chunk and owns
// the dh slice [w*DHT/4, (w+1)*DHT/4) of the output accumulator.
// q/k/v/out are bf16, packed [B, S, nheads*DHT].
// ---------------------------------------------------------------------------
template <int DHT, int MODE>
__global__ __launch_bounds__(128) void k_attn(
    const __bf16* __restrict__ q, const __bf16* __restrict__ k,
    const __bf16* __restrict__ v, __bf16* __restrict__ out,
    const int* __restrict__ xtok, float alpha, int nheads) {
  extern __shared__ char smem[];
  __bf16* Qs   = (__bf16*)smem;            // [16][DHT]
  __bf16* Ks   = Qs + 16 * DHT;            // [64][DHT]
  __bf16* Vt   = Ks + 64 * DHT;            // [DHT][64] (transposed)
  float*  Ss   = (float*)(Vt + 64 * DHT);  // [16][64]
  __bf16* Ps   = (__bf16*)(Ss + 16 * 64);  // [16][64]
  float*  rowM = (float*)(Ps + 16 * 64);
  float*  rowL = rowM + 16;
  float*  rowS = rowL + 16;

  const int tid = threadIdx.x, lane = tid & 31, w = tid >> 5;
  const int lh = lane >> 4, l16 = lane & 15;
  const int s0   = blockIdx.x * 16;
  const int head = blockIdx.y;
  const int bb   = blockIdx.z;
  const int ld   = nheads * DHT;
  const size_t base = (size_t)bb * CS * ld + (size_t)head * DHT;

  // Stage Q tile (async DMA into LDS)
  for (int c = tid; c < 16 * DHT / 8; c += 128) {
    int r = c / (DHT / 8), d = (c % (DHT / 8)) * 8;
    cp_async16(&Qs[r * DHT + d], q + base + (size_t)(s0 + r) * ld + d);
  }
  if (tid < 16) { rowM[tid] = -1e30f; rowL[tid] = 0.0f; }

  constexpr int NT = DHT / 64;   // output 16x16 tiles per wave
  v8f acc[NT];
  #pragma unroll
  for (int nt = 0; nt < NT; ++nt) acc[nt] = v8f{};

  int kstart = 0, kend = CS;
  if (MODE == 1) {
    int a0 = s0 - CW; if (a0 < 0) a0 = 0;
    kstart = a0 & ~63;
    kend = s0 + 16 + CW; if (kend > CS) kend = CS;
  }
  async_fence();
  __syncthreads();

  for (int kb = kstart; kb < kend; kb += 64) {
    // ---- stage K chunk row-major (async) and V chunk transposed (VALU) ----
    for (int c = tid; c < 64 * DHT / 8; c += 128) {
      int r = c / (DHT / 8), d = (c % (DHT / 8)) * 8;
      int kp = kb + r;
      if (kp < CS) {
        cp_async16(&Ks[r * DHT + d], k + base + (size_t)kp * ld + d);
        BF8U u;
        u.f4 = *(const float4*)(v + base + (size_t)kp * ld + d);
        #pragma unroll
        for (int i = 0; i < 8; ++i) Vt[(d + i) * 64 + r] = u.h[i];
      } else {
        *(float4*)(&Ks[r * DHT + d]) = float4{};
        #pragma unroll
        for (int i = 0; i < 8; ++i) Vt[(d + i) * 64 + r] = (__bf16)0.0f;
      }
    }
    async_fence();
    __syncthreads();

    // ---- scores: wave w -> keys [w*16, w*16+16) of this chunk ----
    v8f sacc = v8f{};
    #pragma unroll
    for (int kd = 0; kd < DHT / 32; ++kd) {
      bf16x16 af  = frag_ld(&Qs[l16 * DHT + kd * 32 + lh * 8]);
      bf16x16 bfk = frag_ld(&Ks[(w * 16 + l16) * DHT + kd * 32 + lh * 8]);
      sacc = wmma_bf16(af, bfk, sacc);
    }
    #pragma unroll
    for (int r = 0; r < 8; ++r)
      Ss[(lh * 8 + r) * 64 + w * 16 + l16] = sacc[r];
    __syncthreads();

    // ---- online softmax stats: one thread per query row ----
    if (tid < 16) {
      const int row = tid, qpos = s0 + row;
      float mx = rowM[row];
      for (int c2 = 0; c2 < 64; ++c2) {
        int kp = kb + c2;
        float sv = Ss[row * 64 + c2] * alpha;
        bool ok;
        if (MODE == 0) ok = (xtok[(size_t)bb * CS + kp] != 0);
        else { int dl = kp - qpos; ok = (dl <= CW) && (dl >= -CW) && (kp < CS); }
        if (!ok) sv = -1e9f;
        mx = fmaxf(mx, sv);
      }
      float scl = __expf(rowM[row] - mx);
      float lsum = 0.0f;
      for (int c2 = 0; c2 < 64; ++c2) {
        int kp = kb + c2;
        float sv = Ss[row * 64 + c2] * alpha;
        bool ok;
        if (MODE == 0) ok = (xtok[(size_t)bb * CS + kp] != 0);
        else { int dl = kp - qpos; ok = (dl <= CW) && (dl >= -CW) && (kp < CS); }
        if (!ok) sv = -1e9f;
        float p = __expf(sv - mx);
        lsum += p;
        Ps[row * 64 + c2] = (__bf16)p;
      }
      rowM[row] = mx;
      rowL[row] = rowL[row] * scl + lsum;
      rowS[row] = scl;
    }
    __syncthreads();

    // ---- rescale running output, then O += P * V via WMMA ----
    float rsc[8];
    #pragma unroll
    for (int r = 0; r < 8; ++r) rsc[r] = rowS[lh * 8 + r];
    #pragma unroll
    for (int nt = 0; nt < NT; ++nt)
      #pragma unroll
      for (int r = 0; r < 8; ++r) acc[nt][r] *= rsc[r];

    #pragma unroll
    for (int kk = 0; kk < 64; kk += 32) {
      bf16x16 pf = frag_ld(&Ps[l16 * 64 + kk + lh * 8]);
      #pragma unroll
      for (int nt = 0; nt < NT; ++nt) {
        bf16x16 vf =
            frag_ld(&Vt[(w * (DHT / 4) + nt * 16 + l16) * 64 + kk + lh * 8]);
        acc[nt] = wmma_bf16(pf, vf, acc[nt]);
      }
    }
    __syncthreads();
  }

  // ---- normalize and store ----
  #pragma unroll
  for (int nt = 0; nt < NT; ++nt) {
    #pragma unroll
    for (int r = 0; r < 8; ++r) {
      int row = lh * 8 + r;
      float inv = 1.0f / rowL[row];
      int dh = w * (DHT / 4) + nt * 16 + l16;
      out[base + (size_t)(s0 + row) * ld + dh] = (__bf16)(acc[nt][r] * inv);
    }
  }
}

// ---------------------------------------------------------------------------
// Max-pool over sequence: out[b,d] = max_s a[b,s,d]
// ---------------------------------------------------------------------------
__global__ __launch_bounds__(256) void k_maxpool(const float* __restrict__ a,
                                                 float* __restrict__ o) {
  int idx = blockIdx.x * 256 + threadIdx.x;
  if (idx >= CB * CD) return;
  int bb = idx / CD, d = idx % CD;
  float m = -1e30f;
  for (int s = 0; s < CS; ++s)
    m = fmaxf(m, a[((size_t)bb * CS + s) * CD + d]);
  o[idx] = m;
}

// ---------------------------------------------------------------------------
// Host orchestration
// ---------------------------------------------------------------------------
extern "C" void kernel_launch(void* const* d_in, const int* in_sizes, int n_in,
                              void* d_out, int out_size, void* d_ws,
                              size_t ws_size, hipStream_t stream) {
  (void)in_sizes; (void)n_in; (void)out_size; (void)ws_size;

  const int*   x      = (const int*)d_in[0];
  const float* emb    = (const float*)d_in[1];
  const float* pos    = (const float*)d_in[2];
  const float* ln_e_g = (const float*)d_in[3];
  const float* ln_e_b = (const float*)d_in[4];
  const float* lf_wq  = (const float*)d_in[5];
  const float* lf_wk  = (const float*)d_in[6];
  const float* lf_wv  = (const float*)d_in[7];
  const float* lf_wo  = (const float*)d_in[8];
  const float* ln1_g  = (const float*)d_in[9];
  const float* ln1_b  = (const float*)d_in[10];
  const float* w1     = (const float*)d_in[11];
  const float* w2     = (const float*)d_in[12];
  const float* ln2_g  = (const float*)d_in[13];
  const float* ln2_b  = (const float*)d_in[14];
  const float* mwq    = (const float*)d_in[15];
  const float* mwk    = (const float*)d_in[16];
  const float* mwv    = (const float*)d_in[17];
  const float* mfc    = (const float*)d_in[18];
  const float* mln_g  = (const float*)d_in[19];
  const float* mln_b  = (const float*)d_in[20];

  char* ws = (char*)d_ws;
  size_t off = 0;
  auto alloc = [&](size_t bytes) -> void* {
    void* p = ws + off;
    off += (bytes + 255) & ~(size_t)255;
    return p;
  };

  const size_t BSD = (size_t)CBS * CD;
  const size_t BSF = (size_t)CBS * CDFF;

  float*  hf   = (float*)alloc(BSD * 4);
  __bf16* hb   = (__bf16*)alloc(BSD * 2);
  float*  h1f  = (float*)alloc(BSD * 4);
  __bf16* h1b  = (__bf16*)alloc(BSD * 2);
  float*  olff = (float*)alloc(BSD * 4);
  __bf16* olfb = (__bf16*)alloc(BSD * 2);
  float*  oaf  = (float*)alloc(BSD * 4);
  float*  t1   = (float*)alloc(BSD * 4);
  __bf16* qb   = (__bf16*)alloc(BSD * 2);
  __bf16* kb   = (__bf16*)alloc(BSD * 2);
  __bf16* vb   = (__bf16*)alloc(BSD * 2);
  __bf16* ab   = (__bf16*)alloc(BSD * 2);
  __bf16* f1b  = (__bf16*)alloc(BSF * 2);
  __bf16* mqb  = (__bf16*)alloc(BSF * 2);
  __bf16* mkb  = (__bf16*)alloc(BSF * 2);
  __bf16* mvb  = (__bf16*)alloc(BSF * 2);
  __bf16* mob  = (__bf16*)alloc(BSF * 2);
  __bf16* wqb  = (__bf16*)alloc((size_t)CD * CD * 2);
  __bf16* wkb  = (__bf16*)alloc((size_t)CD * CD * 2);
  __bf16* wvb  = (__bf16*)alloc((size_t)CD * CD * 2);
  __bf16* wob  = (__bf16*)alloc((size_t)CD * CD * 2);
  __bf16* w1b  = (__bf16*)alloc((size_t)CD * CDFF * 2);
  __bf16* w2b  = (__bf16*)alloc((size_t)CDFF * CD * 2);
  __bf16* mwqb = (__bf16*)alloc((size_t)CD * CDFF * 2);
  __bf16* mwkb = (__bf16*)alloc((size_t)CD * CDFF * 2);
  __bf16* mwvb = (__bf16*)alloc((size_t)CD * CDFF * 2);
  __bf16* mfcb = (__bf16*)alloc((size_t)CDFF * CD * 2);

  auto cvt = [&](const float* s, __bf16* d, size_t n) {
    k_cvt<<<dim3((unsigned)((n + 255) / 256)), 256, 0, stream>>>(s, d, (int)n);
  };
  auto gemm = [&](const __bf16* A, const __bf16* B, float* C, __bf16* Cb,
                  int M, int N, int K, int fuse) {
    if (N % 256 == 0 && N >= 1024) {
      k_gemm_bf16<256><<<dim3(N / 256, M / 128), 256, 0, stream>>>(A, B, C, Cb,
                                                                   M, N, K, fuse);
    } else {
      k_gemm_bf16<128><<<dim3(N / 128, M / 128), 256, 0, stream>>>(A, B, C, Cb,
                                                                   M, N, K, fuse);
    }
  };

  // Weight conversion
  cvt(lf_wq, wqb, (size_t)CD * CD);
  cvt(lf_wk, wkb, (size_t)CD * CD);
  cvt(lf_wv, wvb, (size_t)CD * CD);
  cvt(lf_wo, wob, (size_t)CD * CD);
  cvt(w1,  w1b,  (size_t)CD * CDFF);
  cvt(w2,  w2b,  (size_t)CDFF * CD);
  cvt(mwq, mwqb, (size_t)CD * CDFF);
  cvt(mwk, mwkb, (size_t)CD * CDFF);
  cvt(mwv, mwvb, (size_t)CD * CDFF);
  cvt(mfc, mfcb, (size_t)CDFF * CD);

  // h = LN(emb[x] + pos)
  k_embed_ln<<<dim3(CBS), 256, 0, stream>>>(x, emb, pos, ln_e_g, ln_e_b, hf, hb);

  // Longformer QKV projections
  gemm(hb, wqb, nullptr, qb, CBS, CD, CD, 0);
  gemm(hb, wkb, nullptr, kb, CBS, CD, CD, 0);
  gemm(hb, wvb, nullptr, vb, CBS, CD, CD, 0);

  // Sliding-window attention (12 heads, DH=64, band W=256)
  {
    size_t sm = (size_t)(16 + 64 + 64) * 64 * 2 + 16 * 64 * 4 + 16 * 64 * 2 + 3 * 16 * 4;
    k_attn<64, 1><<<dim3(CS / 16, 12, CB), 128, sm, stream>>>(
        qb, kb, vb, ab, nullptr, 0.125f /* 1/sqrt(64) */, 12);
  }

  // Output projection + residual LN
  gemm(ab, wob, t1, nullptr, CBS, CD, CD, 0);
  k_add_ln<<<dim3(CBS), 256, 0, stream>>>(hf, t1, ln1_g, ln1_b, h1f, h1b);

  // FFN with fused GELU
  gemm(h1b, w1b, nullptr, f1b, CBS, CDFF, CD, 1);
  gemm(f1b, w2b, t1, nullptr, CBS, CD, CDFF, 0);
  k_add_ln<<<dim3(CBS), 256, 0, stream>>>(h1f, t1, ln2_g, ln2_b, olff, olfb);

  // Outer MHA projections (NH=4, DK=768)
  gemm(olfb, mwqb, nullptr, mqb, CBS, CDFF, CD, 0);
  gemm(olfb, mwkb, nullptr, mkb, CBS, CDFF, CD, 0);
  gemm(olfb, mwvb, nullptr, mvb, CBS, CDFF, CD, 0);

  // Full attention with key-padding mask
  {
    size_t sm = (size_t)(16 + 64 + 64) * 768 * 2 + 16 * 64 * 4 + 16 * 64 * 2 + 3 * 16 * 4;
    k_attn<768, 0><<<dim3(CS / 16, 4, CB), 128, sm, stream>>>(
        mqb, mkb, mvb, mob, x, 0.03608439182435161f /* 1/sqrt(768) */, 4);
  }

  // fc + residual LN + max-pool
  gemm(mob, mfcb, t1, nullptr, CBS, CD, CDFF, 0);
  k_add_ln<<<dim3(CBS), 256, 0, stream>>>(t1, olff, mln_g, mln_b, oaf, nullptr);
  k_maxpool<<<dim3((CB * CD + 255) / 256), 256, 0, stream>>>(oaf, (float*)d_out);
}
